// Nystrom_MSA_84310208020915
// MI455X (gfx1250) — compile-verified, matrix operand order checked by reading, not run on hardware
//
#include <hip/hip_runtime.h>
#include <hip/hip_bf16.h>

typedef __bf16 bf16_t;
typedef __attribute__((ext_vector_type(16))) __bf16 v16bf;
typedef __attribute__((ext_vector_type(8)))  __bf16 v8bf;
typedef __attribute__((ext_vector_type(8)))  float  v8f;

// ---------------------------------------------------------------------------
// Problem constants
// ---------------------------------------------------------------------------
#define SEQ    4096
#define DMODEL 512
#define NHEAD  8
#define HDIM   64
#define NLMK   256
#define LTOK   16
#define KERN   33
#define QSCALE 0.125f   // 64^-0.5

__device__ __forceinline__ v8f wmma_bf16(v16bf a, v16bf b, v8f c) {
  // emits v_wmma_f32_16x16x32_bf16
  return __builtin_amdgcn_wmma_f32_16x16x32_bf16(
      /*neg_a=*/false, a, /*neg_b=*/false, b,
      /*c_mod=*/(short)0, c, /*reuse_a=*/false, /*reuse_b=*/false);
}

#define SHUF16(lo, hi) __builtin_shufflevector(lo, hi, 0,1,2,3,4,5,6,7,8,9,10,11,12,13,14,15)

// ---------------------------------------------------------------------------
// Batched bf16 GEMM (NT only):  C[m,n] = alpha * sum_k A[m,k] * B[n,k]
//   A: [batch][M][K] row-major bf16 (K-contiguous)
//   B: [batch][N][K] row-major bf16 (K-contiguous) -- i.e. op(B) = B^T
//   C (f32, optional), Cbf (bf16, optional): [batch][M][N]
// Block = 128 threads (4 waves), block tile 128x64, wave tile 32x64,
// K-step 32, double-buffered LDS (one barrier per k-step).
// Per k-step per wave: 12 ds_load_b128 -> 8 v_wmma (B fragment reused 2x).
// Requires M%128==0, N%64==0, K%32==0 (true for all calls below).
// ---------------------------------------------------------------------------
__global__ __launch_bounds__(128)
void gemm_nt_kernel(const bf16_t* __restrict__ A, const bf16_t* __restrict__ B,
                    float* __restrict__ C, bf16_t* __restrict__ Cbf,
                    int M, int N, int K,
                    long long sA, long long sB, long long sC, float alpha)
{
  __shared__ v8bf As[2][128][5];  // [buf][row][k-chunk], chunks 0..3 = K 0..31
  __shared__ v8bf Bs[2][64][5];   // [buf][col][k-chunk]

  const int tid  = threadIdx.x;
  const int lane = tid & 31;
  const int wave = tid >> 5;

  A += (long long)blockIdx.z * sA;
  B += (long long)blockIdx.z * sB;
  const long long cBase = (long long)blockIdx.z * sC;

  const int m0 = blockIdx.y * 128;
  const int n0 = blockIdx.x * 64;

  // staging: A = 512 16B-chunks (4/thread), B = 256 16B-chunks (2/thread)
  const int rs = tid >> 2, cs = tid & 3;

  auto stage = [&](int buf, int k0) {
    #pragma unroll
    for (int i = 0; i < 4; ++i)
      As[buf][rs + 32 * i][cs] =
          *(const v8bf*)(A + (long long)(m0 + rs + 32 * i) * K + (k0 + 8 * cs));
    #pragma unroll
    for (int i = 0; i < 2; ++i)
      Bs[buf][rs + 32 * i][cs] =
          *(const v8bf*)(B + (long long)(n0 + rs + 32 * i) * K + (k0 + 8 * cs));
  };

  v8f acc[2][4];
  #pragma unroll
  for (int u = 0; u < 2; ++u)
    #pragma unroll
    for (int t = 0; t < 4; ++t)
      acc[u][t] = (v8f){0.f,0.f,0.f,0.f,0.f,0.f,0.f,0.f};

  const int g     = lane >> 4;           // K-group of this lane
  const int mbase = wave * 32 + (lane & 15);

  stage(0, 0);

  for (int k0 = 0; k0 < K; k0 += 32) {
    __syncthreads();
    const int cur = (k0 >> 5) & 1;
    if (k0 + 32 < K) {
      stage(cur ^ 1, k0 + 32);           // overlap next-tile staging with WMMA
      __builtin_prefetch(A + (long long)(m0 + rs) * K + (k0 + 64), 0, 1);
      __builtin_prefetch(B + (long long)(n0 + rs) * K + (k0 + 64), 0, 1);
    }
    // A fragments for the wave's two 16-row sub-tiles
    const v16bf a0 = SHUF16(As[cur][mbase][g],      As[cur][mbase][2 + g]);
    const v16bf a1 = SHUF16(As[cur][mbase + 16][g], As[cur][mbase + 16][2 + g]);
    #pragma unroll
    for (int t = 0; t < 4; ++t) {
      const int nc  = t * 16 + (lane & 15);
      const v16bf b = SHUF16(Bs[cur][nc][g], Bs[cur][nc][2 + g]);
      acc[0][t] = wmma_bf16(a0, b, acc[0][t]);
      acc[1][t] = wmma_bf16(a1, b, acc[1][t]);
    }
  }

  // ---- store: lane L -> N = L%16 (+16t), VGPR r -> M = r + 8*(L/16) ----
  #pragma unroll
  for (int u = 0; u < 2; ++u) {
    #pragma unroll
    for (int t = 0; t < 4; ++t) {
      const int col = n0 + t * 16 + (lane & 15);
      #pragma unroll
      for (int r = 0; r < 8; ++r) {
        const int row = m0 + wave * 32 + u * 16 + r + 8 * (lane >> 4);
        const float val = alpha * acc[u][t][r];
        const long long idx = cBase + (long long)row * N + col;
        if (C)   C[idx]   = val;
        if (Cbf) Cbf[idx] = (bf16_t)val;
      }
    }
  }
}

// ---------------------------------------------------------------------------
// Tiled transposes (coalesced both sides)
// ---------------------------------------------------------------------------
// out[c*R + r] = (bf16) in[r*C + c]
__global__ __launch_bounds__(256)
void transpose_f32_to_bf16_kernel(const float* __restrict__ in, bf16_t* __restrict__ out,
                                  int R, int C, long long sIn, long long sOut) {
  __shared__ float tile[32][33];
  in  += (long long)blockIdx.z * sIn;
  out += (long long)blockIdx.z * sOut;
  const int r0 = blockIdx.y * 32, c0 = blockIdx.x * 32;
  const int tr = threadIdx.x >> 5, tc = threadIdx.x & 31;
  #pragma unroll
  for (int i = 0; i < 4; ++i)
    tile[tr + 8 * i][tc] = in[(long long)(r0 + tr + 8 * i) * C + c0 + tc];
  __syncthreads();
  #pragma unroll
  for (int i = 0; i < 4; ++i)
    out[(long long)(c0 + tr + 8 * i) * R + r0 + tc] = (bf16_t)tile[tc][tr + 8 * i];
}

__global__ __launch_bounds__(256)
void transpose_bf16_kernel(const bf16_t* __restrict__ in, bf16_t* __restrict__ out,
                           int R, int C, long long sIn, long long sOut) {
  __shared__ bf16_t tile[32][34];
  in  += (long long)blockIdx.z * sIn;
  out += (long long)blockIdx.z * sOut;
  const int r0 = blockIdx.y * 32, c0 = blockIdx.x * 32;
  const int tr = threadIdx.x >> 5, tc = threadIdx.x & 31;
  #pragma unroll
  for (int i = 0; i < 4; ++i)
    tile[tr + 8 * i][tc] = in[(long long)(r0 + tr + 8 * i) * C + c0 + tc];
  __syncthreads();
  #pragma unroll
  for (int i = 0; i < 4; ++i)
    out[(long long)(c0 + tr + 8 * i) * R + r0 + tc] = tile[tc][tr + 8 * i];
}

// ---------------------------------------------------------------------------
// Elementwise / reduction helpers
// ---------------------------------------------------------------------------
// LayerNorm: one wave32 per row of 512
__global__ __launch_bounds__(256)
void layernorm_kernel(const float* __restrict__ x, const float* __restrict__ g,
                      const float* __restrict__ b, bf16_t* __restrict__ xn) {
  const int row  = blockIdx.x * 8 + (threadIdx.x >> 5);
  const int lane = threadIdx.x & 31;
  if (row >= SEQ) return;
  const float* xr = x + (long long)row * DMODEL;
  float s = 0.f, s2 = 0.f;
  for (int i = lane; i < DMODEL; i += 32) { float v = xr[i]; s += v; s2 += v * v; }
  for (int o = 16; o; o >>= 1) { s += __shfl_xor(s, o, 32); s2 += __shfl_xor(s2, o, 32); }
  const float mu  = s / DMODEL;
  const float var = s2 / DMODEL - mu * mu;
  const float inv = rsqrtf(var + 1e-5f);
  for (int i = lane; i < DMODEL; i += 32)
    xn[(long long)row * DMODEL + i] = (bf16_t)((xr[i] - mu) * inv * g[i] + b[i]);
}

// qkv [N][3*512] f32 -> q,k,v [H][N][HDIM] bf16 (q pre-scaled)
__global__ __launch_bounds__(256)
void split_qkv_kernel(const float* __restrict__ qkv, bf16_t* __restrict__ q,
                      bf16_t* __restrict__ k, bf16_t* __restrict__ v) {
  const int idx = blockIdx.x * 256 + threadIdx.x;   // SEQ*DMODEL
  if (idx >= SEQ * DMODEL) return;
  const int n = idx >> 9, hd = idx & 511, h = hd >> 6, d = hd & 63;
  const long long src = (long long)n * (3 * DMODEL);
  const long long dst = ((long long)h << 18) + ((long long)n << 6) + d;
  q[dst] = (bf16_t)(qkv[src + hd] * QSCALE);
  k[dst] = (bf16_t)(qkv[src + 512 + hd]);
  v[dst] = (bf16_t)(qkv[src + 1024 + hd]);
}

// landmark means over l=16 tokens
__global__ __launch_bounds__(256)
void landmark_mean_kernel(const bf16_t* __restrict__ q, const bf16_t* __restrict__ k,
                          bf16_t* __restrict__ ql, bf16_t* __restrict__ kl) {
  const int idx = blockIdx.x * 256 + threadIdx.x;   // H*NLMK*HDIM
  if (idx >= NHEAD * NLMK * HDIM) return;
  const int d = idx & 63, m = (idx >> 6) & 255, h = idx >> 14;
  const long long base = ((long long)h << 18) + ((long long)(m * LTOK) << 6) + d;
  float sq = 0.f, sk = 0.f;
  #pragma unroll
  for (int j = 0; j < LTOK; ++j) {
    sq += (float)q[base + ((long long)j << 6)];
    sk += (float)k[base + ((long long)j << 6)];
  }
  ql[idx] = (bf16_t)(sq * (1.f / LTOK));
  kl[idx] = (bf16_t)(sk * (1.f / LTOK));
}

// softmax over rows of length 256 (one wave32 / row), f32 in -> bf16 out
__global__ __launch_bounds__(256)
void softmax256_kernel(const float* __restrict__ in, bf16_t* __restrict__ out, int rows) {
  const int row  = blockIdx.x * 8 + (threadIdx.x >> 5);
  const int lane = threadIdx.x & 31;
  if (row >= rows) return;
  const float* r = in + (long long)row * 256;
  float v[8], mx = -1e30f;
  #pragma unroll
  for (int i = 0; i < 8; ++i) { v[i] = r[lane + 32 * i]; mx = fmaxf(mx, v[i]); }
  for (int o = 16; o; o >>= 1) mx = fmaxf(mx, __shfl_xor(mx, o, 32));
  float s = 0.f;
  #pragma unroll
  for (int i = 0; i < 8; ++i) { v[i] = __expf(v[i] - mx); s += v[i]; }
  for (int o = 16; o; o >>= 1) s += __shfl_xor(s, o, 32);
  const float inv = 1.f / s;
  #pragma unroll
  for (int i = 0; i < 8; ++i)
    out[(long long)row * 256 + lane + 32 * i] = (bf16_t)(v[i] * inv);
}

// softmax over rows of length 4096 (one 256-thread block / row)
__global__ __launch_bounds__(256)
void softmax4096_kernel(const float* __restrict__ in, bf16_t* __restrict__ out) {
  __shared__ float red[8];
  const int row = blockIdx.x, tid = threadIdx.x, lane = tid & 31, wv = tid >> 5;
  const float* r = in + (long long)row * SEQ;
  float v[16], mx = -1e30f;
  #pragma unroll
  for (int i = 0; i < 16; ++i) { v[i] = r[tid + 256 * i]; mx = fmaxf(mx, v[i]); }
  for (int o = 16; o; o >>= 1) mx = fmaxf(mx, __shfl_xor(mx, o, 32));
  if (lane == 0) red[wv] = mx;
  __syncthreads();
  float m2 = red[lane & 7];
  for (int o = 4; o; o >>= 1) m2 = fmaxf(m2, __shfl_xor(m2, o, 32));
  mx = m2;
  float s = 0.f;
  #pragma unroll
  for (int i = 0; i < 16; ++i) { v[i] = __expf(v[i] - mx); s += v[i]; }
  for (int o = 16; o; o >>= 1) s += __shfl_xor(s, o, 32);
  __syncthreads();
  if (lane == 0) red[wv] = s;
  __syncthreads();
  float s2 = red[lane & 7];
  for (int o = 4; o; o >>= 1) s2 += __shfl_xor(s2, o, 32);
  const float inv = 1.f / s2;
  #pragma unroll
  for (int i = 0; i < 16; ++i)
    out[(long long)row * SEQ + tid + 256 * i] = (bf16_t)(v[i] * inv);
}

// ---- Newton-Schulz pinv helpers (x = attn2 bf16, [H][256][256]) ----
__global__ void zero2_kernel(unsigned int* s) {
  if (blockIdx.x == 0 && threadIdx.x < 2) s[threadIdx.x] = 0u;
}
__global__ __launch_bounds__(256)
void absmax_rows_kernel(const bf16_t* __restrict__ x, unsigned int* s) {
  const int idx = blockIdx.x * 256 + threadIdx.x;    // H*256 rows
  if (idx >= NHEAD * NLMK) return;
  const bf16_t* r = x + (long long)idx * NLMK;
  float sum = 0.f;
  for (int j = 0; j < NLMK; ++j) sum += fabsf((float)r[j]);
  atomicMax(&s[0], __float_as_uint(sum));
}
__global__ __launch_bounds__(256)
void absmax_cols_kernel(const bf16_t* __restrict__ x, unsigned int* s) {
  const int idx = blockIdx.x * 256 + threadIdx.x;    // H*256 cols
  if (idx >= NHEAD * NLMK) return;
  const int h = idx >> 8, c = idx & 255;
  const bf16_t* p = x + ((long long)h << 16) + c;
  float sum = 0.f;
  for (int j = 0; j < NLMK; ++j) sum += fabsf((float)p[(long long)j * NLMK]);
  atomicMax(&s[1], __float_as_uint(sum));
}
// z0 = x^T / denom (row-major) AND z0T = x / denom
__global__ __launch_bounds__(256)
void pinv_zinit_kernel(const bf16_t* __restrict__ x, bf16_t* __restrict__ z,
                       bf16_t* __restrict__ zT, const unsigned int* __restrict__ s) {
  const long long idx = (long long)blockIdx.x * 256 + threadIdx.x;  // H*256*256
  if (idx >= (long long)NHEAD * NLMK * NLMK) return;
  const int h = (int)(idx >> 16), r = (int)((idx >> 8) & 255), c = (int)(idx & 255);
  const float inv = 1.f / (__uint_as_float(s[0]) * __uint_as_float(s[1]));
  z[idx]  = (bf16_t)((float)x[((long long)h << 16) + ((long long)c << 8) + r] * inv);
  zT[idx] = (bf16_t)((float)x[idx] * inv);
}
// outT = bf16( (diagC*I - in)^T ) : outT[h,n,k] = diagC*d(n,k) - in[h,k,n]
__global__ __launch_bounds__(256)
void diag_sub_t_kernel(const float* __restrict__ in, bf16_t* __restrict__ outT, float diagC) {
  const long long idx = (long long)blockIdx.x * 256 + threadIdx.x;  // H*256*256
  if (idx >= (long long)NHEAD * NLMK * NLMK) return;
  const int h = (int)(idx >> 16), n = (int)((idx >> 8) & 255), k = (int)(idx & 255);
  outT[idx] = (bf16_t)((n == k ? diagC : 0.f) -
                       in[((long long)h << 16) + ((long long)k << 8) + n]);
}

// out_heads[h,n,d] += depthwise_conv(v)[h,n,d]; write proj_in bf16 [n][h*64+d]
__global__ __launch_bounds__(256)
void conv_combine_kernel(const float* __restrict__ out_heads, const bf16_t* __restrict__ v,
                         const float* __restrict__ conv_w, bf16_t* __restrict__ proj_in) {
  const int idx = blockIdx.x * 256 + threadIdx.x;    // H*SEQ*HDIM
  if (idx >= NHEAD * SEQ * HDIM) return;
  const int d = idx & 63, n = (idx >> 6) & (SEQ - 1), h = idx >> 18;
  float acc = out_heads[idx];
  const long long vh = (long long)h << 18;
  #pragma unroll 1
  for (int j = 0; j < KERN; ++j) {
    const int p = n + j - (KERN / 2);
    if (p >= 0 && p < SEQ)
      acc += conv_w[h * KERN + j] * (float)v[vh + ((long long)p << 6) + d];
  }
  proj_in[(long long)n * DMODEL + h * HDIM + d] = (bf16_t)acc;
}

// out = proj + b_out ; xo = x + out
__global__ __launch_bounds__(256)
void final_epilogue_kernel(const float* __restrict__ proj, const float* __restrict__ bias,
                           const float* __restrict__ x, float* __restrict__ out_xo,
                           float* __restrict__ out_o) {
  const long long idx = (long long)blockIdx.x * 256 + threadIdx.x;  // SEQ*DMODEL
  if (idx >= (long long)SEQ * DMODEL) return;
  const float v = proj[idx] + bias[idx & 511];
  out_o[idx]  = v;
  out_xo[idx] = x[idx] + v;
}

// ---------------------------------------------------------------------------
// Host orchestration
// ---------------------------------------------------------------------------
extern "C" void kernel_launch(void* const* d_in, const int* in_sizes, int n_in,
                              void* d_out, int out_size, void* d_ws, size_t ws_size,
                              hipStream_t stream) {
  (void)in_sizes; (void)n_in; (void)out_size; (void)ws_size;
  const float* x      = (const float*)d_in[0];
  const float* ln_g   = (const float*)d_in[1];
  const float* ln_b   = (const float*)d_in[2];
  const float* w_qkv  = (const float*)d_in[3];
  const float* w_out  = (const float*)d_in[4];
  const float* b_out  = (const float*)d_in[5];
  const float* conv_w = (const float*)d_in[6];

  float* o_xout = (float*)d_out;                                   // [4096,512]
  float* o_out  = o_xout + (long long)SEQ * DMODEL;                // [4096,512]
  float* o_attn = o_out  + (long long)SEQ * DMODEL;                // [8,4096,4096]

  // ---- workspace carve-up ----
  char* ws = (char*)d_ws;
  size_t off = 0;
  auto alloc = [&](size_t bytes) -> void* {
    void* p = ws + off;
    off = (off + bytes + 255) & ~(size_t)255;
    return p;
  };
  const long long HND = (long long)NHEAD * SEQ * HDIM;     // 2M
  const long long HMM = (long long)NHEAD * NLMK * NLMK;    // 512K
  const long long HNM = (long long)NHEAD * SEQ * NLMK;     // 8M

  bf16_t* xn_bf     = (bf16_t*)alloc((size_t)SEQ * DMODEL * 2);
  bf16_t* wqkvT_bf  = (bf16_t*)alloc((size_t)DMODEL * 3 * DMODEL * 2);  // [1536][512]
  bf16_t* woutT_bf  = (bf16_t*)alloc((size_t)DMODEL * DMODEL * 2);      // [512][512]
  float*  qkv_f32   = (float*)alloc((size_t)SEQ * 3 * DMODEL * 4);      // reused as out_heads
  bf16_t* q_bf      = (bf16_t*)alloc((size_t)HND * 2);
  bf16_t* k_bf      = (bf16_t*)alloc((size_t)HND * 2);
  bf16_t* v_bf      = (bf16_t*)alloc((size_t)HND * 2);
  bf16_t* vT_bf     = (bf16_t*)alloc((size_t)HND * 2);                  // [H][64][4096]
  bf16_t* ql_bf     = (bf16_t*)alloc((size_t)NHEAD * NLMK * HDIM * 2);
  bf16_t* kl_bf     = (bf16_t*)alloc((size_t)NHEAD * NLMK * HDIM * 2);
  float*  sim1_f32  = (float*)alloc((size_t)HNM * 4);
  float*  sim2_f32  = (float*)alloc((size_t)HMM * 4);
  float*  sim3_f32  = (float*)alloc((size_t)HNM * 4);                   // reused as proj_f32
  bf16_t* attn1_bf  = (bf16_t*)alloc((size_t)HNM * 2);
  bf16_t* attn2_bf  = (bf16_t*)alloc((size_t)HMM * 2);
  bf16_t* attn3_bf  = (bf16_t*)alloc((size_t)HNM * 2);                  // [H][256][4096]
  bf16_t* attn3T_bf = (bf16_t*)alloc((size_t)HNM * 2);                  // [H][4096][256]
  unsigned int* scal = (unsigned int*)alloc(256);
  bf16_t* z_a       = (bf16_t*)alloc((size_t)HMM * 2);
  bf16_t* zT_a      = (bf16_t*)alloc((size_t)HMM * 2);
  bf16_t* z_b       = (bf16_t*)alloc((size_t)HMM * 2);
  bf16_t* zT_b      = (bf16_t*)alloc((size_t)HMM * 2);
  float*  xz_f32    = (float*)alloc((size_t)HMM * 4);
  bf16_t* xz_bf     = (bf16_t*)alloc((size_t)HMM * 2);
  float*  p_f32     = (float*)alloc((size_t)HMM * 4);
  bf16_t* nT_bf     = (bf16_t*)alloc((size_t)HMM * 2);
  bf16_t* tmp2_bf   = (bf16_t*)alloc((size_t)NHEAD * NLMK * HDIM * 2);  // [H][256][64]
  bf16_t* tmp2T_bf  = (bf16_t*)alloc((size_t)NHEAD * NLMK * HDIM * 2);  // [H][64][256]
  bf16_t* tmp1_bf   = (bf16_t*)alloc((size_t)HNM * 2);
  bf16_t* projin_bf = (bf16_t*)alloc((size_t)SEQ * DMODEL * 2);
  float*  out_heads = qkv_f32;    // reuse (qkv dead after split)
  float*  proj_f32  = sim3_f32;   // reuse (sim3 dead after softmax)

  const dim3 blk256(256), blkG(128);
  auto cdiv = [](long long a, long long b) { return (unsigned)((a + b - 1) / b); };

  // 1) weights -> bf16, transposed to K-contiguous [N][K]
  transpose_f32_to_bf16_kernel<<<dim3((3 * DMODEL) / 32, DMODEL / 32, 1), blk256, 0, stream>>>(
      w_qkv, wqkvT_bf, DMODEL, 3 * DMODEL, 0, 0);
  transpose_f32_to_bf16_kernel<<<dim3(DMODEL / 32, DMODEL / 32, 1), blk256, 0, stream>>>(
      w_out, woutT_bf, DMODEL, DMODEL, 0, 0);

  // 2) LayerNorm
  layernorm_kernel<<<SEQ / 8, blk256, 0, stream>>>(x, ln_g, ln_b, xn_bf);

  // 3) QKV GEMM: [4096,512] x [512,1536] (B' = wqkvT [1536][512])
  gemm_nt_kernel<<<dim3((3 * DMODEL) / 64, SEQ / 128, 1), blkG, 0, stream>>>(
      xn_bf, wqkvT_bf, qkv_f32, nullptr, SEQ, 3 * DMODEL, DMODEL, 0, 0, 0, 1.f);

  // 4) split + head-major pack (q scaled)
  split_qkv_kernel<<<cdiv((long long)SEQ * DMODEL, 256), blk256, 0, stream>>>(
      qkv_f32, q_bf, k_bf, v_bf);

  // 5) landmark means
  landmark_mean_kernel<<<cdiv((long long)NHEAD * NLMK * HDIM, 256), blk256, 0, stream>>>(
      q_bf, k_bf, ql_bf, kl_bf);

  // 6) similarity GEMMs (already NT: B' rows are K-contiguous)
  gemm_nt_kernel<<<dim3(NLMK / 64, SEQ / 128, NHEAD), blkG, 0, stream>>>(
      q_bf, kl_bf, sim1_f32, nullptr, SEQ, NLMK, HDIM,
      (long long)SEQ * HDIM, (long long)NLMK * HDIM, (long long)SEQ * NLMK, 1.f);
  gemm_nt_kernel<<<dim3(NLMK / 64, NLMK / 128, NHEAD), blkG, 0, stream>>>(
      ql_bf, kl_bf, sim2_f32, nullptr, NLMK, NLMK, HDIM,
      (long long)NLMK * HDIM, (long long)NLMK * HDIM, (long long)NLMK * NLMK, 1.f);
  gemm_nt_kernel<<<dim3(SEQ / 64, NLMK / 128, NHEAD), blkG, 0, stream>>>(
      ql_bf, k_bf, sim3_f32, nullptr, NLMK, SEQ, HDIM,
      (long long)NLMK * HDIM, (long long)SEQ * HDIM, (long long)NLMK * SEQ, 1.f);

  // 7) softmaxes -> bf16
  softmax256_kernel<<<(NHEAD * SEQ) / 8, blk256, 0, stream>>>(sim1_f32, attn1_bf, NHEAD * SEQ);
  softmax256_kernel<<<(NHEAD * NLMK) / 8, blk256, 0, stream>>>(sim2_f32, attn2_bf, NHEAD * NLMK);
  softmax4096_kernel<<<NHEAD * NLMK, blk256, 0, stream>>>(sim3_f32, attn3_bf);

  // 8) K-contiguous copies for later NT GEMMs
  transpose_bf16_kernel<<<dim3(SEQ / 32, NLMK / 32, NHEAD), blk256, 0, stream>>>(
      attn3_bf, attn3T_bf, NLMK, SEQ, (long long)NLMK * SEQ, (long long)NLMK * SEQ);
  transpose_bf16_kernel<<<dim3(HDIM / 32, SEQ / 32, NHEAD), blk256, 0, stream>>>(
      v_bf, vT_bf, SEQ, HDIM, (long long)SEQ * HDIM, (long long)SEQ * HDIM);

  // 9) pinv init: z0 = attn2^T / (max rowsum * max colsum), plus z0T
  zero2_kernel<<<1, 32, 0, stream>>>(scal);
  absmax_rows_kernel<<<cdiv(NHEAD * NLMK, 256), blk256, 0, stream>>>(attn2_bf, scal);
  absmax_cols_kernel<<<cdiv(NHEAD * NLMK, 256), blk256, 0, stream>>>(attn2_bf, scal);
  pinv_zinit_kernel<<<cdiv(HMM, 256), blk256, 0, stream>>>(attn2_bf, z_a, zT_a, scal);

  // 10) Newton-Schulz: z = 0.25 * z @ (13I - xz@(15I - xz@(7I - xz)))
  const dim3 gP(NLMK / 64, NLMK / 128, NHEAD);
  const long long sMM = (long long)NLMK * NLMK;
  bf16_t *zc = z_a, *zcT = zT_a, *zn = z_b, *znT = zT_b;
  for (int it = 0; it < 6; ++it) {
    gemm_nt_kernel<<<gP, blkG, 0, stream>>>(            // xz = x @ z   (B' = zT)
        attn2_bf, zcT, xz_f32, xz_bf, NLMK, NLMK, NLMK, sMM, sMM, sMM, 1.f);
    diag_sub_t_kernel<<<cdiv(HMM, 256), blk256, 0, stream>>>(xz_f32, nT_bf, 7.f);
    gemm_nt_kernel<<<gP, blkG, 0, stream>>>(            // t2 = xz @ (7I-xz)
        xz_bf, nT_bf, p_f32, nullptr, NLMK, NLMK, NLMK, sMM, sMM, sMM, 1.f);
    diag_sub_t_kernel<<<cdiv(HMM, 256), blk256, 0, stream>>>(p_f32, nT_bf, 15.f);
    gemm_nt_kernel<<<gP, blkG, 0, stream>>>(            // t4 = xz @ (15I-t2)
        xz_bf, nT_bf, p_f32, nullptr, NLMK, NLMK, NLMK, sMM, sMM, sMM, 1.f);
    diag_sub_t_kernel<<<cdiv(HMM, 256), blk256, 0, stream>>>(p_f32, nT_bf, 13.f);
    gemm_nt_kernel<<<gP, blkG, 0, stream>>>(            // z' = 0.25 z @ (13I-t4)
        zc, nT_bf, nullptr, zn, NLMK, NLMK, NLMK, sMM, sMM, sMM, 0.25f);
    transpose_bf16_kernel<<<dim3(NLMK / 32, NLMK / 32, NHEAD), blk256, 0, stream>>>(
        zn, znT, NLMK, NLMK, sMM, sMM);
    bf16_t* t;
    t = zc; zc = zn; zn = t;
    t = zcT; zcT = znT; znT = t;
  }
  // zc/zcT now hold attn2_inv (row-major / K-contiguous-T)

  // 11) tmp2 = attn3 @ v : [256,4096] x [4096,64]  (B' = vT)
  gemm_nt_kernel<<<dim3(HDIM / 64, NLMK / 128, NHEAD), blkG, 0, stream>>>(
      attn3_bf, vT_bf, nullptr, tmp2_bf, NLMK, HDIM, SEQ,
      (long long)NLMK * SEQ, (long long)SEQ * HDIM, (long long)NLMK * HDIM, 1.f);
  transpose_bf16_kernel<<<dim3(HDIM / 32, NLMK / 32, NHEAD), blk256, 0, stream>>>(
      tmp2_bf, tmp2T_bf, NLMK, HDIM,
      (long long)NLMK * HDIM, (long long)NLMK * HDIM);

  // 12) tmp1 = attn1 @ attn2_inv : [4096,256] x [256,256]  (B' = zcT)
  gemm_nt_kernel<<<dim3(NLMK / 64, SEQ / 128, NHEAD), blkG, 0, stream>>>(
      attn1_bf, zcT, nullptr, tmp1_bf, SEQ, NLMK, NLMK,
      (long long)SEQ * NLMK, sMM, (long long)SEQ * NLMK, 1.f);

  // 13) attn = tmp1 @ attn3 : [4096,256] x [256,4096] -> f32 output (B' = attn3T)
  gemm_nt_kernel<<<dim3(SEQ / 64, SEQ / 128, NHEAD), blkG, 0, stream>>>(
      tmp1_bf, attn3T_bf, o_attn, nullptr, SEQ, SEQ, NLMK,
      (long long)SEQ * NLMK, (long long)NLMK * SEQ, (long long)SEQ * SEQ, 1.f);

  // 14) out_heads = tmp1 @ tmp2 : [4096,256] x [256,64]  (B' = tmp2T)
  gemm_nt_kernel<<<dim3(HDIM / 64, SEQ / 128, NHEAD), blkG, 0, stream>>>(
      tmp1_bf, tmp2T_bf, out_heads, nullptr, SEQ, HDIM, NLMK,
      (long long)SEQ * NLMK, (long long)NLMK * HDIM, (long long)SEQ * HDIM, 1.f);

  // 15) residual depthwise conv + repack to [N, H*DH] bf16
  conv_combine_kernel<<<cdiv(HND, 256), blk256, 0, stream>>>(
      out_heads, v_bf, conv_w, projin_bf);

  // 16) output projection: [4096,512] x [512,512]  (B' = woutT)
  gemm_nt_kernel<<<dim3(DMODEL / 64, SEQ / 128, 1), blkG, 0, stream>>>(
      projin_bf, woutT_bf, proj_f32, nullptr, SEQ, DMODEL, DMODEL, 0, 0, 0, 1.f);

  // 17) bias + residual -> first two outputs
  final_epilogue_kernel<<<cdiv((long long)SEQ * DMODEL, 256), blk256, 0, stream>>>(
      proj_f32, b_out, x, o_xout, o_out);
}
